// binary_decoder_24867860644389
// MI455X (gfx1250) — compile-verified, matrix operand order checked by reading, not run on hardware
//
#include <hip/hip_runtime.h>
#include <hip/hip_bf16.h>
#include <stdint.h>

#define IN_F   4096
#define OUT_F  768
#define NB     8
#define BATCH  4096
#define WCOLS  (OUT_F * NB)   // 6144

#define BM 128
#define BN 64
#define BK 32
#define LDST 40               // padded LDS row stride in halfs (80B) -> conflict-free b128 reads

typedef __attribute__((ext_vector_type(16))) _Float16 v16h;
typedef __attribute__((ext_vector_type(8)))  _Float16 half8;
typedef __attribute__((ext_vector_type(8)))  float    v8f;
typedef __attribute__((ext_vector_type(4)))  float    f32x4;
typedef __attribute__((ext_vector_type(4)))  unsigned int u32x4;

union Frag { v16h v; half8 h[2]; };

#if defined(__gfx1250__) && __has_builtin(__builtin_amdgcn_global_load_async_to_lds_b128) && __has_builtin(__builtin_amdgcn_s_wait_asynccnt)
#define USE_ASYNC 1
#else
#define USE_ASYNC 0
#endif

__device__ __forceinline__ void cp16(_Float16* l, const _Float16* g) {
#if USE_ASYNC
  typedef int v4i __attribute__((vector_size(16)));
  typedef __attribute__((address_space(1))) v4i gv4;   // global
  typedef __attribute__((address_space(3))) v4i lv4;   // LDS
  __builtin_amdgcn_global_load_async_to_lds_b128(
      (gv4*)(uintptr_t)g,   // generic->AS1 is a bit-identical no-op
      (lv4*)l,              // generic->AS3 addrspacecast
      0, 0);
#else
  *(u32x4*)l = *(const u32x4*)g;
#endif
}

// ---------------- kernel 0: zero accumulators ----------------
__global__ void k_init(float* acc) {
  if (threadIdx.x < 2) acc[threadIdx.x] = 0.0f;
}

// ---------------- kernel 1: latent f32 -> f16 ----------------
__global__ __launch_bounds__(256) void k_convert(const float* __restrict__ src,
                                                 _Float16* __restrict__ dst) {
  size_t i = ((size_t)blockIdx.x * 256 + threadIdx.x) * 8;
  f32x4 a = *(const f32x4*)(src + i);
  f32x4 b = *(const f32x4*)(src + i + 4);
  half8 h;
  h[0] = (_Float16)a[0]; h[1] = (_Float16)a[1];
  h[2] = (_Float16)a[2]; h[3] = (_Float16)a[3];
  h[4] = (_Float16)b[0]; h[5] = (_Float16)b[1];
  h[6] = (_Float16)b[2]; h[7] = (_Float16)b[3];
  *(half8*)(dst + i) = h;
}

// ------- kernel 2: sigmoid-decode weights -> Wt[out][in] (f16), polarize loss -------
__global__ __launch_bounds__(256) void k_decode(const float* __restrict__ weight,
                                                _Float16* __restrict__ wt,
                                                float* __restrict__ acc) {
  int t   = blockIdx.x * 256 + threadIdx.x;
  int in  = t & (IN_F - 1);
  int out = t >> 12;                       // t / 4096
  const float* wp = weight + (size_t)in * WCOLS + (size_t)out * NB;
  f32x4 w0 = *(const f32x4*)wp;
  f32x4 w1 = *(const f32x4*)(wp + 4);
  float wv[8] = {w0[0], w0[1], w0[2], w0[3], w1[0], w1[1], w1[2], w1[3]};
  const float pw[8] = {1.f, 2.f, 4.f, 8.f, 16.f, 32.f, 64.f, -128.f};
  float s = 0.f, pol = 0.f;
#pragma unroll
  for (int b = 0; b < 8; ++b) {
    float p = 1.0f / (1.0f + __builtin_expf(-wv[b]));
    s   += p * pw[b];
    pol += p * (1.0f - p);
  }
  wt[(size_t)out * IN_F + in] = (_Float16)s;   // transposed, coalesced store

  __shared__ float red[256];
  red[threadIdx.x] = pol;
  __syncthreads();
#pragma unroll
  for (int st = 128; st > 0; st >>= 1) {
    if ((int)threadIdx.x < st) red[threadIdx.x] += red[threadIdx.x + st];
    __syncthreads();
  }
  if (threadIdx.x == 0) atomicAdd(acc + 1, red[0]);
}

// ------- kernel 3: WMMA GEMM (pred = latent @ Wt^T) fused with recon loss -------
__global__ __launch_bounds__(256) void k_gemm_loss(const _Float16* __restrict__ A,   // [BATCH][IN_F]
                                                   const _Float16* __restrict__ Bt,  // [OUT_F][IN_F]
                                                   const float* __restrict__ tsum,   // [BATCH][WCOLS]
                                                   float* __restrict__ acc) {
  // A rows 0..127 at lds[buf][row*LDST], B rows 0..63 at lds[buf][(128+row)*LDST]
  __shared__ _Float16 lds[2][(BM + BN) * LDST];
  __shared__ float red[256];

  const int tid   = threadIdx.x;
  const int lane  = tid & 31;
  const int w     = tid >> 5;     // 0..7 waves
  const int waveM = w & 3;        // 4 waves along M (32 rows each)
  const int waveN = w >> 2;       // 2 waves along N (32 cols each)
  const int m0    = blockIdx.y * BM;
  const int n0    = blockIdx.x * BN;

  v8f c[2][2];
#pragma unroll
  for (int mt = 0; mt < 2; ++mt)
#pragma unroll
    for (int nt = 0; nt < 2; ++nt)
#pragma unroll
      for (int r = 0; r < 8; ++r) c[mt][nt][r] = 0.0f;

  const int seg   = tid & 3;        // 16B segment within a 64B k-row chunk
  const int rowT  = tid >> 2;       // 0..63

  auto stage = [&](int buf, int k0) {
    // A: 128 rows x 32 halfs (two passes of 64 rows); B: 64 rows
    {
      int row = rowT;                                           // A rows 0..63
      cp16(&lds[buf][row * LDST + (seg << 3)],
           A + (size_t)(m0 + row) * IN_F + k0 + (seg << 3));
    }
    {
      int row = 64 + rowT;                                      // A rows 64..127
      cp16(&lds[buf][row * LDST + (seg << 3)],
           A + (size_t)(m0 + row) * IN_F + k0 + (seg << 3));
    }
    {
      int row = rowT;                                           // B rows 0..63
      cp16(&lds[buf][(BM + row) * LDST + (seg << 3)],
           Bt + (size_t)(n0 + row) * IN_F + k0 + (seg << 3));
    }
  };

  stage(0, 0);
  const int NK = IN_F / BK;   // 128
  const int half = lane >> 4; // 0/1: K-half select per ISA 16-bit A/B layout
  const int r16  = lane & 15;

#pragma unroll 1
  for (int kk = 0; kk < NK; ++kk) {
    int buf = kk & 1;
    if (kk + 1 < NK) {
      stage(buf ^ 1, (kk + 1) * BK);
#if USE_ASYNC
      __builtin_amdgcn_s_wait_asynccnt(3);   // this wave's 3 loads for `buf` are done (in-order)
#endif
    } else {
#if USE_ASYNC
      __builtin_amdgcn_s_wait_asynccnt(0);
#endif
    }
    __syncthreads();

    Frag fa[2], fb[2];
#pragma unroll
    for (int mt = 0; mt < 2; ++mt) {
      int row = waveM * 32 + mt * 16 + r16;
      const _Float16* p = &lds[buf][row * LDST + 8 * half];
      fa[mt].h[0] = *(const half8*)p;        // K = 8*half .. +7
      fa[mt].h[1] = *(const half8*)(p + 16); // K = 16+8*half .. +7
    }
#pragma unroll
    for (int nt = 0; nt < 2; ++nt) {
      int row = BM + waveN * 32 + nt * 16 + r16;
      const _Float16* p = &lds[buf][row * LDST + 8 * half];
      fb[nt].h[0] = *(const half8*)p;
      fb[nt].h[1] = *(const half8*)(p + 16);
    }
#pragma unroll
    for (int mt = 0; mt < 2; ++mt)
#pragma unroll
      for (int nt = 0; nt < 2; ++nt)
        c[mt][nt] = __builtin_amdgcn_wmma_f32_16x16x32_f16(
            false, fa[mt].v, false, fb[nt].v, (short)0, c[mt][nt], false, false);

    __syncthreads();   // all waves done reading `buf` before next stage overwrites it
  }

  // fused epilogue: decode int_sum from true_sum and accumulate squared error
  float lsum = 0.0f;
#pragma unroll
  for (int mt = 0; mt < 2; ++mt)
#pragma unroll
    for (int nt = 0; nt < 2; ++nt)
#pragma unroll
      for (int r = 0; r < 8; ++r) {
        int m = m0 + waveM * 32 + mt * 16 + r + 8 * half;  // C layout: VGPR r, lane halves
        int n = n0 + waveN * 32 + nt * 16 + r16;
        const float* ts = tsum + (size_t)m * WCOLS + (size_t)n * NB;
        f32x4 t0 = *(const f32x4*)ts;
        f32x4 t1 = *(const f32x4*)(ts + 4);
        float is = t0[0] + 2.f * t0[1] + 4.f * t0[2] + 8.f * t0[3]
                 + 16.f * t1[0] + 32.f * t1[1] + 64.f * t1[2] - 128.f * t1[3];
        float d = c[mt][nt][r] - is;
        lsum += d * d;
      }

  red[tid] = lsum;
  __syncthreads();
#pragma unroll
  for (int st = 128; st > 0; st >>= 1) {
    if (tid < st) red[tid] += red[tid + st];
    __syncthreads();
  }
  if (tid == 0) atomicAdd(acc + 0, red[0]);
}

// ---------------- kernel 4: finalize means ----------------
__global__ void k_final(const float* acc, float* out) {
  if (threadIdx.x == 0) {
    out[0] = acc[0] / ((float)BATCH * (float)OUT_F) / 128.0f;  // recon_loss
    out[1] = acc[1] / ((float)IN_F * (float)WCOLS);            // polarize_loss
  }
}

extern "C" void kernel_launch(void* const* d_in, const int* in_sizes, int n_in,
                              void* d_out, int out_size, void* d_ws, size_t ws_size,
                              hipStream_t stream) {
  const float* latent = (const float*)d_in[0];
  const float* tsum   = (const float*)d_in[1];
  const float* weight = (const float*)d_in[2];

  char* ws = (char*)d_ws;
  _Float16* wsLat = (_Float16*)ws;                                   // 32 MiB
  _Float16* wsWt  = (_Float16*)(ws + (size_t)BATCH * IN_F * 2);      //  6 MiB
  float*    acc   = (float*)(ws + (size_t)BATCH * IN_F * 2 + (size_t)OUT_F * IN_F * 2);
  float*    out   = (float*)d_out;

  k_init<<<1, 32, 0, stream>>>(acc);
  k_convert<<<(BATCH * IN_F) / (256 * 8), 256, 0, stream>>>(latent, wsLat);
  k_decode<<<(IN_F * OUT_F) / 256, 256, 0, stream>>>(weight, wsWt, acc);
  k_gemm_loss<<<dim3(OUT_F / BN, BATCH / BM), 256, 0, stream>>>(wsLat, wsWt, tsum, acc);
  k_final<<<1, 32, 0, stream>>>(acc, out);
}
